// MultiHeadAttention_51488067944691
// MI455X (gfx1250) — compile-verified, matrix operand order checked by reading, not run on hardware
//
#include <hip/hip_runtime.h>

#define DEV __device__ __forceinline__

typedef __attribute__((ext_vector_type(16))) __bf16          v16bf;
typedef __attribute__((ext_vector_type(16))) unsigned short  v16u;
typedef __attribute__((ext_vector_type(8)))  float           v8f;
typedef __attribute__((ext_vector_type(4)))  int             v4i;

static constexpr int Bb = 8, Ss = 1024, Dd = 1024, Hh = 16, DKk = 1024, HDd = 64;

// ---------------------------------------------------- async-LDS detection ---
#if defined(__has_builtin)
#  if __has_builtin(__builtin_amdgcn_global_load_async_to_lds_b128) && \
      __has_builtin(__builtin_amdgcn_s_wait_asynccnt)
#    define USE_ASYNC_LDS 1
#  endif
#endif
#ifndef USE_ASYNC_LDS
#  define USE_ASYNC_LDS 0
#endif

#if USE_ASYNC_LDS
typedef __attribute__((address_space(1))) v4i  g_v4i;
typedef __attribute__((address_space(3))) v4i  l_v4i;
DEV void async_b128(const unsigned short* g, unsigned short* l) {
  __builtin_amdgcn_global_load_async_to_lds_b128(
      (g_v4i*)g, (l_v4i*)l, 0, 0);
}
#endif

// ---------------------------------------------------------------- helpers ---
DEV unsigned short f2bf(float f) {
  unsigned int u = __builtin_bit_cast(unsigned int, f);
  u += 0x7FFFu + ((u >> 16) & 1u);          // round-to-nearest-even
  return (unsigned short)(u >> 16);
}

DEV v16bf asbf(v16u u) { return __builtin_bit_cast(v16bf, u); }

DEV v8f wmma_bf16(v16u a, v16u b, v8f c) {
  return __builtin_amdgcn_wmma_f32_16x16x32_bf16(
      false, asbf(a), false, asbf(b), (short)0, c, false, false);
}

// 16-bit A/B fragment for one 16x16x32 step. `row` points at a 32-element,
// 16B-aligned K-run. Lane-half kh: elems 0..7 -> K=kh*8+0..7, 8..15 -> K=16+kh*8+0..7.
DEV v16u frag16(const unsigned short* row, int kh) {
  union { uint4 u[2]; v16u v; } x;
  x.u[0] = *(const uint4*)(row + kh * 8);
  x.u[1] = *(const uint4*)(row + 16 + kh * 8);
  return x.v;
}

DEV uint4 pack8bf(v8f v) {                   // 8 f32 -> 8 bf16 packed in a b128
  union { unsigned short s[8]; uint4 u; } x;
#pragma unroll
  for (int r = 0; r < 8; ++r) x.s[r] = f2bf(v[r]);
  return x.u;
}

// ------------------------------------------------------------- conversion ---
__global__ __launch_bounds__(256) void cvt_f32_bf16(const float* __restrict__ in,
                                                    unsigned short* __restrict__ out,
                                                    int n) {
  int i = (blockIdx.x * blockDim.x + threadIdx.x) * 4;
  if (i + 3 < n) {
    float4 f = *(const float4*)(in + i);
    out[i + 0] = f2bf(f.x);
    out[i + 1] = f2bf(f.y);
    out[i + 2] = f2bf(f.z);
    out[i + 3] = f2bf(f.w);
  } else {
    for (; i < n; ++i) out[i] = f2bf(in[i]);
  }
}

// Transpose + convert: in (R x C) f32 -> out (C x R) bf16, batched over z.
__global__ __launch_bounds__(256) void cvt_t_f32_bf16(const float* __restrict__ in,
                                                      unsigned short* __restrict__ out,
                                                      int R, int C) {
  __shared__ float tile[32][33];
  const int z = blockIdx.z;
  in  += (long long)z * R * C;
  out += (long long)z * R * C;
  const int tx = threadIdx.x & 31, ty = threadIdx.x >> 5;   // 32x8
  const int r0 = blockIdx.y * 32, c0 = blockIdx.x * 32;
#pragma unroll
  for (int i = 0; i < 4; ++i)
    tile[ty + i * 8][tx] = in[(long long)(r0 + ty + i * 8) * C + c0 + tx];
  __syncthreads();
#pragma unroll
  for (int i = 0; i < 4; ++i)
    out[(long long)(c0 + ty + i * 8) * R + r0 + tx] = f2bf(tile[tx][ty + i * 8]);
}

// ------------------------------------------------------------- NT GEMM ------
// C(MxN) = A(MxK,bf16 rowmajor) * Bt(NxK,bf16 rowmajor)^T, double-buffered LDS.
// Block 256 (8 waves, 2x4 grid), tile 128x128, k-step 32, 4x2 frags per wave.
// Staging: async global->LDS b128 when available, else regs double-buffer.
// Store modes: 0 = bf16 rowmajor, 1 = bf16 transposed (+z*nOffPerZ N-offset,
// packed b128), 2 = bf16 per-head transposed v_t[b][h][d][s] (packed b128),
// 3 = f32 rowmajor + bias.
__global__ __launch_bounds__(256) void gemm_bf16_nt(
    const unsigned short* __restrict__ A, const unsigned short* __restrict__ Bt,
    void* __restrict__ Cv, const float* __restrict__ bias,
    int M, int N, int K, int lda, int ldb, int ldc,
    long long strideA, long long strideB, int nOffPerZ, int mode) {
  __shared__ __align__(16) unsigned short As[2][128 * 32];
  __shared__ __align__(16) unsigned short Bs[2][128 * 32];

  const int z = blockIdx.z;
  A  += (long long)z * strideA;
  Bt += (long long)z * strideB;

  const int tid  = threadIdx.x;
  const int wave = tid >> 5, lane = tid & 31;
  const int wr = wave >> 2, wc = wave & 3;
  const int ml = lane & 15, kh = lane >> 4, mh = kh * 8;
  const int m0 = blockIdx.y * 128, n0 = blockIdx.x * 128;

#if USE_ASYNC_LDS
  auto aload = [&](const unsigned short* P, int ld, int r0, int lim, int k0,
                   unsigned short* S) {
#pragma unroll
    for (int v = 0; v < 2; ++v) {
      int idx = tid + v * 256;                 // 512 vectors of 8 bf16 = 128x32
      int r = idx >> 2, c = (idx & 3) * 8;
      unsigned short* l = S + r * 32 + c;
      if (r0 + r < lim)
        async_b128(P + (long long)(r0 + r) * ld + k0 + c, l);
      else
        *(uint4*)l = make_uint4(0, 0, 0, 0);
    }
  };
#else
  auto gload = [&](const unsigned short* P, int ld, int r0, int lim, int k0,
                   uint4* out) {
#pragma unroll
    for (int v = 0; v < 2; ++v) {
      int idx = tid + v * 256;
      int r = idx >> 2, c = (idx & 3) * 8;
      uint4 val = make_uint4(0, 0, 0, 0);
      if (r0 + r < lim) val = *(const uint4*)(P + (long long)(r0 + r) * ld + k0 + c);
      out[v] = val;
    }
  };
  auto sstore = [&](unsigned short* S, const uint4* in) {
#pragma unroll
    for (int v = 0; v < 2; ++v) {
      int idx = tid + v * 256;
      int r = idx >> 2, c = (idx & 3) * 8;
      *(uint4*)(S + r * 32 + c) = in[v];
    }
  };
#endif

  v8f acc[4][2] = {};

#if USE_ASYNC_LDS
  aload(A, lda, m0, M, 0, As[0]);
  aload(Bt, ldb, n0, N, 0, Bs[0]);
  __builtin_amdgcn_s_wait_asynccnt(0);
  __syncthreads();
#else
  uint4 ra[2], rb[2];
  gload(A, lda, m0, M, 0, ra);
  gload(Bt, ldb, n0, N, 0, rb);
  sstore(As[0], ra);
  sstore(Bs[0], rb);
  __syncthreads();
#endif

  const int nk = K >> 5;
  for (int ki = 0; ki < nk; ++ki) {
    const int cur = ki & 1, nxt = cur ^ 1;
    const bool more = (ki + 1) < nk;
#if USE_ASYNC_LDS
    if (more) {                                // async DMA next tile into LDS
      aload(A, lda, m0, M, (ki + 1) * 32, As[nxt]);
      aload(Bt, ldb, n0, N, (ki + 1) * 32, Bs[nxt]);
    }
#else
    if (more) {                                // prefetch next tile into regs
      gload(A, lda, m0, M, (ki + 1) * 32, ra);
      gload(Bt, ldb, n0, N, (ki + 1) * 32, rb);
    }
#endif
    v16u au[4], bu[2];
#pragma unroll
    for (int i = 0; i < 4; ++i)
      au[i] = frag16(&As[cur][(wr * 64 + i * 16 + ml) * 32], kh);
#pragma unroll
    for (int j = 0; j < 2; ++j)
      bu[j] = frag16(&Bs[cur][(wc * 32 + j * 16 + ml) * 32], kh);
#pragma unroll
    for (int i = 0; i < 4; ++i)
#pragma unroll
      for (int j = 0; j < 2; ++j)
        acc[i][j] = wmma_bf16(au[i], bu[j], acc[i][j]);
#if USE_ASYNC_LDS
    if (more) __builtin_amdgcn_s_wait_asynccnt(0);
#else
    if (more) {
      sstore(As[nxt], ra);
      sstore(Bs[nxt], rb);
    }
#endif
    __syncthreads();
  }

#pragma unroll
  for (int i = 0; i < 4; ++i)
#pragma unroll
    for (int j = 0; j < 2; ++j) {
      const int gmB = m0 + wr * 64 + i * 16 + mh;   // 8 consecutive rows gmB..+7
      const int gn  = n0 + wc * 32 + j * 16 + ml;
      if (gn >= N) continue;
      if (mode == 0) {
        unsigned short* C = (unsigned short*)Cv;
#pragma unroll
        for (int r = 0; r < 8; ++r)
          if (gmB + r < M) C[(long long)(gmB + r) * ldc + gn] = f2bf(acc[i][j][r]);
      } else if (mode == 1) {
        unsigned short* C = (unsigned short*)Cv;
        *(uint4*)(C + (long long)(z * nOffPerZ + gn) * ldc + gmB) = pack8bf(acc[i][j]);
      } else if (mode == 2) {
        unsigned short* C = (unsigned short*)Cv;
        long long addr =
            ((long long)((gmB >> 10) * Hh + (gn >> 6)) * 64 + (gn & 63)) * Ss +
            (gmB & 1023);
        *(uint4*)(C + addr) = pack8bf(acc[i][j]);
      } else {
        float* C = (float*)Cv;
        float bv = bias ? bias[gn] : 0.f;
#pragma unroll
        for (int r = 0; r < 8; ++r)
          if (gmB + r < M) C[(long long)(gmB + r) * ldc + gn] = acc[i][j][r] + bv;
      }
    }
}

// -------------------------------------------- attention pass 1: col stats ---
// softmax over the QUERY axis: per key column k, m[k]=max_q s, l[k]=sum_q e^(s-m).
// Grid (S/128, H, B); 8 waves x 16 columns; kT B-frags live in regs across q loop.
__global__ __launch_bounds__(256) void attn_stats(
    const unsigned short* __restrict__ qg, const unsigned short* __restrict__ kg,
    float* __restrict__ col_m, float* __restrict__ col_l) {
  const int b = blockIdx.z, h = blockIdx.y, c0 = blockIdx.x * 128;
  const int tid = threadIdx.x, wave = tid >> 5, lane = tid & 31;
  const int ml = lane & 15, kh = lane >> 4;
  const int colBase = c0 + wave * 16;
  __shared__ __align__(16) unsigned short qs[16 * 64];

  const unsigned short* krow =
      kg + ((long long)(b * Ss + colBase + ml)) * DKk + h * 64;
  v16u bu[2];                                  // kT frags: B[d][n] = k[colBase+n][d]
  bu[0] = frag16(krow, kh);
  bu[1] = frag16(krow + 32, kh);

  float m_run = -INFINITY, l_run = 0.f;
  const float scale = 0.03125f;                // 1/sqrt(1024)

  for (int q0 = 0; q0 < Ss; q0 += 16) {
    {                                          // stage 16x64 q tile
      int idx = tid * 4, r = idx >> 6, c = idx & 63;
      *(uint2*)(&qs[r * 64 + c]) =
          *(const uint2*)(&qg[((long long)(b * Ss + q0 + r)) * DKk + h * 64 + c]);
    }
    __syncthreads();

    v8f sc = {};
    sc = wmma_bf16(frag16(&qs[ml * 64], kh), bu[0], sc);
    sc = wmma_bf16(frag16(&qs[ml * 64 + 32], kh), bu[1], sc);

    float sv[8], tmax = -INFINITY;
#pragma unroll
    for (int r = 0; r < 8; ++r) { sv[r] = sc[r] * scale; tmax = fmaxf(tmax, sv[r]); }
    tmax = fmaxf(tmax, __shfl_xor(tmax, 16, 32));  // merge half-rows (same column)
    float mnew = fmaxf(m_run, tmax);
    float psum = 0.f;
#pragma unroll
    for (int r = 0; r < 8; ++r) psum += __expf(sv[r] - mnew);
    psum += __shfl_xor(psum, 16, 32);
    l_run = l_run * __expf(m_run - mnew) + psum;
    m_run = mnew;
    __syncthreads();
  }

  if (lane < 16) {
    long long idx = (long long)(b * Hh + h) * Ss + colBase + lane;
    col_m[idx] = m_run;
    col_l[idx] = l_run;
  }
}

// --------------------------------- attention pass 2: apply + attn@V fused ---
// Grid (S/128, H, B). Block computes 128x64 output rows; streams keys in
// 64-chunks, recomputes score tile, writes bf16 attn to LDS, contracts with V.
// V comes in d-major layout v_t[b][h][d][s] so its fragments are contiguous.
__global__ __launch_bounds__(256) void attn_apply_av(
    const unsigned short* __restrict__ qg, const unsigned short* __restrict__ kg,
    const unsigned short* __restrict__ vt, const float* __restrict__ col_m,
    const float* __restrict__ col_l, unsigned short* __restrict__ og) {
  const int b = blockIdx.z, h = blockIdx.y, q0 = blockIdx.x * 128;
  const int tid = threadIdx.x, wave = tid >> 5, lane = tid & 31;
  const int ml = lane & 15, kh = lane >> 4, mh = kh * 8;

  __shared__ __align__(16) unsigned short qs[128 * 64];    // [q][d]   16 KB
  __shared__ __align__(16) unsigned short ks_[64 * 64];    // [k][d]    8 KB
  __shared__ __align__(16) unsigned short vs_t[64 * 64];   // [d][k]    8 KB
  __shared__ __align__(16) unsigned short as_[128 * 64];   // [q][k]   16 KB

#pragma unroll
  for (int v = 0; v < 4; ++v) {                // stage 128x64 q tile
    int idx = (tid + v * 256) * 8, r = idx >> 6, c = idx & 63;
    *(uint4*)(&qs[r * 64 + c]) =
        *(const uint4*)(&qg[((long long)(b * Ss + q0 + r)) * DKk + h * 64 + c]);
  }

  v8f oacc[4] = {};
  const float scale = 0.03125f;
  const long long mlBase = (long long)(b * Hh + h) * Ss;
  const unsigned short* vtb = vt + (long long)((b * Hh + h) * 64) * Ss;

  for (int kc = 0; kc < Ss; kc += 64) {
#pragma unroll
    for (int v = 0; v < 2; ++v) {              // stage k chunk [k][d], vT chunk [d][k]
      int idx = (tid + v * 256) * 8, r = idx >> 6, c = idx & 63;
      *(uint4*)(&ks_[r * 64 + c]) =
          *(const uint4*)(&kg[((long long)(b * Ss + kc + r)) * DKk + h * 64 + c]);
      *(uint4*)(&vs_t[r * 64 + c]) =
          *(const uint4*)(&vtb[(long long)r * Ss + kc + c]);
    }
    __syncthreads();

    // phase A: s = q @ kT for this wave's 16 rows, 4 column tiles
    v16u aq[2];
    aq[0] = frag16(&qs[(wave * 16 + ml) * 64], kh);
    aq[1] = frag16(&qs[(wave * 16 + ml) * 64 + 32], kh);
#pragma unroll
    for (int t = 0; t < 4; ++t) {
      v8f sacc = {};
      sacc = wmma_bf16(aq[0], frag16(&ks_[(t * 16 + ml) * 64], kh), sacc);
      sacc = wmma_bf16(aq[1], frag16(&ks_[(t * 16 + ml) * 64 + 32], kh), sacc);

      long long cg = mlBase + kc + t * 16 + ml;     // this lane's key column
      float mm = col_m[cg];
      float rl = 1.0f / col_l[cg];
#pragma unroll
      for (int r = 0; r < 8; ++r) {
        float a = __expf(sacc[r] * scale - mm) * rl;
        as_[(wave * 16 + mh + r) * 64 + t * 16 + ml] = f2bf(a);
      }
    }
    __syncthreads();

    // phase B: oacc += attnChunk(128x64) @ v(64x64)
    v16u aa[2];
    aa[0] = frag16(&as_[(wave * 16 + ml) * 64], kh);
    aa[1] = frag16(&as_[(wave * 16 + ml) * 64 + 32], kh);
#pragma unroll
    for (int t = 0; t < 4; ++t) {
      oacc[t] = wmma_bf16(aa[0], frag16(&vs_t[(t * 16 + ml) * 64], kh), oacc[t]);
      oacc[t] = wmma_bf16(aa[1], frag16(&vs_t[(t * 16 + ml) * 64 + 32], kh), oacc[t]);
    }
    __syncthreads();
  }

#pragma unroll
  for (int t = 0; t < 4; ++t)
#pragma unroll
    for (int r = 0; r < 8; ++r) {
      int gm = q0 + wave * 16 + mh + r;
      og[((long long)(b * Ss + gm)) * DKk + h * 64 + t * 16 + ml] = f2bf(oacc[t][r]);
    }
}

// -------------------------------------------------------------- host side ---
extern "C" void kernel_launch(void* const* d_in, const int* in_sizes, int n_in,
                              void* d_out, int out_size, void* d_ws, size_t ws_size,
                              hipStream_t stream) {
  (void)in_sizes; (void)n_in; (void)out_size; (void)ws_size;
  const float* src = (const float*)d_in[0];
  const float* tgt = (const float*)d_in[1];
  const float* wq  = (const float*)d_in[2];
  const float* wk  = (const float*)d_in[3];
  const float* wv  = (const float*)d_in[4];
  const float* WQ  = (const float*)d_in[5];
  const float* WK  = (const float*)d_in[6];
  const float* WV  = (const float*)d_in[7];
  const float* ow  = (const float*)d_in[8];
  const float* ob  = (const float*)d_in[9];

  char* ws = (char*)d_ws;
  size_t off = 0;
  auto take = [&](size_t bytes) -> char* {
    char* p = ws + off;
    off = (off + bytes + 255) & ~(size_t)255;
    return p;
  };
  const size_t nAct = (size_t)Bb * Ss * Dd;   // 8M elems
  const size_t nW   = (size_t)Dd * DKk;       // 1M elems

  unsigned short* src_h = (unsigned short*)take(nAct * 2);
  unsigned short* tgt_h = (unsigned short*)take(nAct * 2);
  unsigned short* wq_h  = (unsigned short*)take(nW * 2);
  unsigned short* wk_h  = (unsigned short*)take(nW * 2);
  unsigned short* wv_h  = (unsigned short*)take(nW * 2);
  unsigned short* WQt   = (unsigned short*)take(nW * 2);  // (H, HD, D)
  unsigned short* WKt   = (unsigned short*)take(nW * 2);
  unsigned short* WVt   = (unsigned short*)take(nW * 2);
  unsigned short* owt   = (unsigned short*)take(nW * 2);  // (D, DK) = ow^T
  unsigned short* Wcqt  = (unsigned short*)take(nW * 2);  // (DK, D) = (wq@WQ)^T
  unsigned short* Wckt  = (unsigned short*)take(nW * 2);
  unsigned short* Wcvt  = (unsigned short*)take(nW * 2);
  unsigned short* q_all = (unsigned short*)take(nAct * 2);
  unsigned short* k_all = (unsigned short*)take(nAct * 2);
  unsigned short* v_t   = (unsigned short*)take(nAct * 2);  // [b][h][d][s]
  unsigned short* o_all = (unsigned short*)take(nAct * 2);
  float* col_m = (float*)take((size_t)Bb * Hh * Ss * 4);
  float* col_l = (float*)take((size_t)Bb * Hh * Ss * 4);

  dim3 blk(256);
  auto cvt = [&](const float* in, unsigned short* out, size_t n) {
    cvt_f32_bf16<<<dim3((unsigned)((n / 4 + 255) / 256)), blk, 0, stream>>>(in, out, (int)n);
  };
  cvt(src, src_h, nAct);
  cvt(tgt, tgt_h, nAct);
  cvt(wq, wq_h, nW);
  cvt(wk, wk_h, nW);
  cvt(wv, wv_h, nW);
  // transposed bf16 weights
  cvt_t_f32_bf16<<<dim3(HDd / 32, Dd / 32, Hh), blk, 0, stream>>>(WQ, WQt, Dd, HDd);
  cvt_t_f32_bf16<<<dim3(HDd / 32, Dd / 32, Hh), blk, 0, stream>>>(WK, WKt, Dd, HDd);
  cvt_t_f32_bf16<<<dim3(HDd / 32, Dd / 32, Hh), blk, 0, stream>>>(WV, WVt, Dd, HDd);
  cvt_t_f32_bf16<<<dim3(Dd / 32, DKk / 32, 1), blk, 0, stream>>>(ow, owt, DKk, Dd);

  auto gemm = [&](const unsigned short* A, const unsigned short* Bt, void* C,
                  const float* bias, int M, int N, int K, int lda, int ldb, int ldc,
                  long long sA, long long sB, int nOffPerZ, int nb, int mode) {
    dim3 g((N + 127) / 128, (M + 127) / 128, nb);
    gemm_bf16_nt<<<g, blk, 0, stream>>>(A, Bt, C, bias, M, N, K, lda, ldb, ldc,
                                        sA, sB, nOffPerZ, mode);
  };

  // Weight combine (batched over H), output TRANSPOSED: Wct[h*64+j][d] = (w@W[h])[d][j]
  gemm(wq_h, WQt, Wcqt, nullptr, Dd, HDd, DKk, DKk, Dd, Dd,
       0, (long long)HDd * Dd, HDd, Hh, 1);
  gemm(wk_h, WKt, Wckt, nullptr, Dd, HDd, DKk, DKk, Dd, Dd,
       0, (long long)HDd * Dd, HDd, Hh, 1);
  gemm(wv_h, WVt, Wcvt, nullptr, Dd, HDd, DKk, DKk, Dd, Dd,
       0, (long long)HDd * Dd, HDd, Hh, 1);

  // Fused projections (8192 x 1024 x 1024): C = act @ Wc = act @ (Wct)^T
  gemm(src_h, Wcqt, q_all, nullptr, Bb * Ss, DKk, Dd, Dd, Dd, DKk, 0, 0, 0, 1, 0);
  gemm(tgt_h, Wckt, k_all, nullptr, Bb * Ss, DKk, Dd, Dd, Dd, DKk, 0, 0, 0, 1, 0);
  gemm(tgt_h, Wcvt, v_t,   nullptr, Bb * Ss, DKk, Dd, Dd, Dd, DKk, 0, 0, 0, 1, 2);

  // Attention (softmax over query axis): column stats, then fused apply+AV
  dim3 ga(Ss / 128, Hh, Bb);
  attn_stats<<<ga, blk, 0, stream>>>(q_all, k_all, col_m, col_l);
  attn_apply_av<<<ga, blk, 0, stream>>>(q_all, k_all, v_t, col_m, col_l, o_all);

  // Output projection: out = o_all @ ow + ob = o_all @ (owt)^T + ob  (f32)
  gemm(o_all, owt, d_out, ob, Bb * Ss, Dd, DKk, DKk, DKk, Dd, 0, 0, 0, 1, 3);
}
